// NNConv_49177375539505
// MI455X (gfx1250) — matrix operand order repaired and checked by reading, not compiled, not run on hardware
//
#include <hip/hip_runtime.h>
#include <hip/hip_bf16.h>

// ---------------------------------------------------------------------------
// NNConv on MI455X (gfx1250), bf16 WMMA (v_wmma_f32_16x16x32_bf16), wave32.
//
//   h       = relu(edge_attr @ W1 + b1)            [E,128]
//   weights = (h @ W2 + b2) -> [E,16,16]
//   msg[e]  = src[e] (1x16) @ weights[e] (16x16)
//   out     = scatter_add(msg, dst) + node_attr @ root
//
// Fusion: N-tile t of (h @ W2) is row i=t of the per-edge weight matrix
// (col = i*16 + o), so msg accumulates as  msg += src[:,t] * tile_t  without
// materializing the [E,256] weights tensor. b1 is folded into the layer-1
// WMMA via the zero-padded K=16 slot (A gets 1.0, B gets b1), b2 via one
// extra WMMA (src_tile @ reshape(b2,16,16)) -> all accumulators start as
// SRC2 inline 0. W2 (64KB bf16 fragments) staged in LDS with async loads.
// Each wave runs TWO 16-edge tiles so every W2 B-fragment LDS load feeds
// two WMMAs (DS:WMMA ratio halved; W2 global traffic ~200MB total).
// ---------------------------------------------------------------------------

#define N_NODES 50000
#define N_EDGES 800000
#define IN_CH   16
#define OUT_CH  16
#define EDGE_DIM 16
#define HIDDEN  128

typedef __attribute__((ext_vector_type(16))) __bf16 v16bf;
typedef __attribute__((ext_vector_type(8)))  float  v8f;

// ---------------------------------------------------------------------------
// K0: pre-swizzle weights (f32) into bf16 WMMA B-fragment layout in ws.
// B-fragment (32x16 bf16, 16 elems/lane): n = lane%16,
//   element j -> k = (lane<16 ? 0 : 8) + (j&7) + (j<8 ? 0 : 16)   (within the 32-wide K window)
// W1 fragments additionally carry b1 in row k==16 (bias-through-matmul).
// ---------------------------------------------------------------------------
__global__ void build_frags(const float* __restrict__ W1,
                            const float* __restrict__ W2,
                            const float* __restrict__ b1,
                            const float* __restrict__ b2,
                            __bf16* __restrict__ w1f,     // [8 frag][32 lane][16]
                            __bf16* __restrict__ w2f,     // [64 frag][32 lane][16]
                            __bf16* __restrict__ b2f) {   // [32 lane][16]
    int tid = blockIdx.x * blockDim.x + threadIdx.x;
    if (tid < 8 * 32) {                       // W1 + b1: 8 N-tiles
        int lane = tid & 31;
        int t    = tid >> 5;
        int n    = t * 16 + (lane & 15);
        #pragma unroll
        for (int j = 0; j < 16; ++j) {
            int k = ((lane & 16) ? 8 : 0) + (j & 7) + ((j & 8) ? 16 : 0);
            float v = (k < EDGE_DIM) ? W1[k * HIDDEN + n]
                                     : ((k == EDGE_DIM) ? b1[n] : 0.0f);
            w1f[(size_t)tid * 16 + j] = (__bf16)v;
        }
    } else if (tid < 8 * 32 + 64 * 32) {      // W2: frag = t*4 + kk
        int id   = tid - 256;
        int lane = id & 31;
        int frag = id >> 5;
        int t = frag >> 2, kk = frag & 3;
        int n = t * 16 + (lane & 15);
        #pragma unroll
        for (int j = 0; j < 16; ++j) {
            int k = kk * 32 + ((lane & 16) ? 8 : 0) + (j & 7) + ((j & 8) ? 16 : 0);
            w2f[(size_t)id * 16 + j] = (__bf16)W2[k * (IN_CH * OUT_CH) + n];
        }
    } else if (tid < 8 * 32 + 64 * 32 + 32) { // b2 as a 16x16 B-tile (K pad 32)
        int lane = tid - 2304;
        int n    = lane & 15;
        #pragma unroll
        for (int j = 0; j < 16; ++j) {
            int k = ((lane & 16) ? 8 : 0) + (j & 7) + ((j & 8) ? 16 : 0);
            float v = (k < 16) ? b2[k * 16 + n] : 0.0f;
            b2f[(size_t)lane * 16 + j] = (__bf16)v;
        }
    }
}

// ---------------------------------------------------------------------------
// K1: out = node_attr @ root  (initializes poisoned d_out; 25 MF, negligible)
// ---------------------------------------------------------------------------
__global__ void root_transform(const float* __restrict__ node_attr,
                               const float* __restrict__ root,
                               float* __restrict__ out) {
    int n = blockIdx.x * blockDim.x + threadIdx.x;
    if (n >= N_NODES) return;
    float x[16];
    #pragma unroll
    for (int i = 0; i < 16; ++i) x[i] = node_attr[(size_t)n * 16 + i];
    #pragma unroll
    for (int o = 0; o < 16; ++o) {
        float acc = 0.0f;
        #pragma unroll
        for (int i = 0; i < 16; ++i) acc += x[i] * root[i * 16 + o];
        out[(size_t)n * 16 + o] = acc;
    }
}

// ---------------------------------------------------------------------------
// K2: fused edge MLP + message + scatter. 1 wave = 2 x 16 edges, 8 waves/block.
// ---------------------------------------------------------------------------
__global__ __launch_bounds__(256, 1) void nnconv_edges(
        const float* __restrict__ node_attr,
        const float* __restrict__ edge_attr,
        const int*   __restrict__ edge_index,   // [2][E]: row0=src, row1=dst
        const __bf16* __restrict__ w1f,
        const __bf16* __restrict__ w2f,
        const __bf16* __restrict__ b2f,
        float* __restrict__ out) {

    __shared__ __bf16 lds_w2[64 * 32 * 16];        // 64 KB: W2 B-fragments
    __shared__ __bf16 lds_h[8][2][4 * 32 * 16];    // 64 KB: h in A-frag layout
    __shared__ float  lds_src[8][2][16 * 16];      // 16 KB: gathered src rows

    const int tid  = threadIdx.x;
    const int lane = tid & 31;
    const int wave = tid >> 5;
    const int mrow = lane & 15;                 // n (B/C/D) or m (A) = lane%16
    const int hi   = (lane >> 4) & 1;           // lane half -> M/K offset +8

    // --- stage W2 fragments into LDS: async global->LDS, 16B per lane x16 ---
    {
        unsigned ldsbase = (unsigned)(unsigned long long)&lds_w2[0]; // LDS offset
        #pragma unroll
        for (int i = 0; i < 16; ++i) {
            unsigned loff = ldsbase + (unsigned)(tid + 256 * i) * 16u;
            unsigned long long gaddr =
                (unsigned long long)(w2f) + (unsigned long long)(tid + 256 * i) * 16ull;
            asm volatile("global_load_async_to_lds_b128 %0, %1, off"
                         :: "v"(loff), "v"(gaddr) : "memory");
        }
        asm volatile("s_wait_asynccnt 0" ::: "memory");
    }
    __syncthreads();

    const int base = (blockIdx.x * 8 + wave) * 32;   // 32 edges per wave

    // --- per-tile: gather src rows (regs + LDS) and build A1 fragments ---
    v16bf a_src[2], a1[2];
    #pragma unroll
    for (int u = 0; u < 2; ++u) {
        const int e0 = base + u * 16;
        int srcNode = edge_index[e0 + mrow];
        const float* p = node_attr + (size_t)srcNode * 16 + hi * 8;
        float*       q = &lds_src[wave][u][mrow * 16 + hi * 8];
        float srow[8];
        #pragma unroll
        for (int j = 0; j < 8; ++j) { srow[j] = p[j]; q[j] = srow[j]; }
        #pragma unroll
        for (int j = 0; j < 8; ++j) a_src[u][j] = (__bf16)srow[j];
        #pragma unroll
        for (int j = 8; j < 16; ++j) a_src[u][j] = (__bf16)0.0f;

        // edge_attr tile 16x16, K padded to 32; K=16 slot = 1.0 so the
        // W1-fragment row k==16 (=b1) adds the bias in-WMMA.
        const float* ea = edge_attr + (size_t)(e0 + mrow) * 16;
        #pragma unroll
        for (int j = 0; j < 8; ++j) a1[u][j] = (__bf16)ea[hi * 8 + j];
        #pragma unroll
        for (int j = 9; j < 16; ++j) a1[u][j] = (__bf16)0.0f;
        a1[u][8] = hi ? (__bf16)0.0f : (__bf16)1.0f;   // K=16 (lanes 0-15)
    }

    // --- layer 1: h = relu(A1 @ [W1;b1]), C = inline 0; store relu(h) in
    //     A-fragment layout so layer-2 A loads are contiguous b128s. ---
    for (int t1 = 0; t1 < 8; ++t1) {
        v16bf bw1 = *(const v16bf*)(w1f + ((size_t)t1 * 32 + lane) * 16);
        // C elem r is h[m = r + hi*8, kglob = t1*16 + mrow] -> slot (lane2, j)
        // of fragment kk = t1>>1 per the 16-bit A layout.
        int kk   = t1 >> 1;
        int hi2  = (mrow >> 3) & 1;
        int j    = (mrow & 7) + ((t1 & 1) << 3);
        int off  = kk * 512 + (hi * 8 + hi2 * 16) * 16 + j;
        #pragma unroll
        for (int u = 0; u < 2; ++u) {
            v8f c = {};
            c = __builtin_amdgcn_wmma_f32_16x16x32_bf16(false, a1[u], false, bw1,
                                                        (short)0, c, false, false);
            #pragma unroll
            for (int r = 0; r < 8; ++r) {
                float v = __builtin_fmaxf(c[r], 0.0f);   // single v_max_num_f32
                lds_h[wave][u][off + r * 16] = (__bf16)v;
            }
        }
    }
    __syncthreads();   // h + src visible

    // --- A2 fragments: 4 contiguous 32B loads per lane per tile ---
    v16bf a2[2][4];
    #pragma unroll
    for (int u = 0; u < 2; ++u)
        #pragma unroll
        for (int kk = 0; kk < 4; ++kk)
            a2[u][kk] = *(const v16bf*)&lds_h[wave][u][kk * 512 + lane * 16];

    // --- msg starts as the b2 bias term: src_tile @ reshape(b2,16,16) ---
    v8f msg[2];
    {
        v16bf bb2 = *(const v16bf*)(b2f + (size_t)lane * 16);
        #pragma unroll
        for (int u = 0; u < 2; ++u) {
            v8f m0 = {};
            msg[u] = __builtin_amdgcn_wmma_f32_16x16x32_bf16(false, a_src[u], false, bb2,
                                                             (short)0, m0, false, false);
        }
    }

    // --- fused layer 2 + message contraction; each B-fragment load feeds
    //     two WMMAs (one per edge-tile). ---
    for (int t = 0; t < 16; ++t) {              // t = input-channel row of W
        v8f acc[2] = {{}, {}};                  // SRC2 inline 0 on first WMMA
        #pragma unroll
        for (int kk = 0; kk < 4; ++kk) {
            v16bf bw2 = *(const v16bf*)&lds_w2[((t * 4 + kk) * 32 + lane) * 16];
            acc[0] = __builtin_amdgcn_wmma_f32_16x16x32_bf16(false, a2[0][kk], false, bw2,
                                                             (short)0, acc[0], false, false);
            acc[1] = __builtin_amdgcn_wmma_f32_16x16x32_bf16(false, a2[1][kk], false, bw2,
                                                             (short)0, acc[1], false, false);
        }
        #pragma unroll
        for (int u = 0; u < 2; ++u)
            #pragma unroll
            for (int r = 0; r < 8; ++r) {       // msg[e,:] += src[e,t] * row_t
                float s = lds_src[wave][u][(r + hi * 8) * 16 + t];
                msg[u][r] += s * acc[u][r];
            }
    }

    // --- scatter-add to destination nodes ---
    #pragma unroll
    for (int u = 0; u < 2; ++u)
        #pragma unroll
        for (int r = 0; r < 8; ++r) {
            int m = r + hi * 8;
            int d = edge_index[N_EDGES + base + u * 16 + m];
            atomicAdd(&out[(size_t)d * 16 + mrow], msg[u][r]);
        }
}

// ---------------------------------------------------------------------------
extern "C" void kernel_launch(void* const* d_in, const int* in_sizes, int n_in,
                              void* d_out, int out_size, void* d_ws, size_t ws_size,
                              hipStream_t stream) {
    const float* node_attr  = (const float*)d_in[0];
    const float* edge_attr  = (const float*)d_in[1];
    const float* W1         = (const float*)d_in[2];
    const float* b1         = (const float*)d_in[3];
    const float* W2         = (const float*)d_in[4];
    const float* b2         = (const float*)d_in[5];
    const float* root       = (const float*)d_in[6];
    const int*   edge_index = (const int*)d_in[7];
    float* out = (float*)d_out;

    __bf16* w1f = (__bf16*)d_ws;                          // 8 KB
    __bf16* w2f = (__bf16*)((char*)d_ws + 8192);          // 64 KB
    __bf16* b2f = (__bf16*)((char*)d_ws + 8192 + 65536);  // 1 KB

    build_frags<<<10, 256, 0, stream>>>(W1, W2, b1, b2, w1f, w2f, b2f);
    root_transform<<<(N_NODES + 255) / 256, 256, 0, stream>>>(node_attr, root, out);
    nnconv_edges<<<N_EDGES / (16 * 2 * 8), 256, 0, stream>>>(
        node_attr, edge_attr, edge_index, w1f, w2f, b2f, out);
}
